// hetero_actor_19705309954765
// MI455X (gfx1250) — matrix operand order repaired and checked by reading, not compiled, not run on hardware
//
#include <hip/hip_runtime.h>
#include <math.h>

typedef __attribute__((ext_vector_type(16))) _Float16 v16h;
typedef __attribute__((ext_vector_type(8)))  float    v8f;
typedef __attribute__((ext_vector_type(4)))  float    vf4;

static constexpr int   kNT     = 100000;
static constexpr int   kNJ     = 300000;
static constexpr int   kE_TJ   = 1000000;
static constexpr int   kE_JT   = 1000000;
static constexpr int   kE_JJ   = 2000000;
static constexpr float kSPBias = 0.54132485461291810f;  // log(expm1(1.0))

// ---------------------------------------------------------------------------
// Pack f32 weights [K x Ncols] (row-major) into a contiguous-k WMMA f16 B-operand
// lane layout. Layout per 512-half block (kc, nt): index = ((kc*Ntiles+nt)*32
// + lane)*16 + i, where element i of lane's v16h holds W[K][N] with
// K = kc*32 + i + 16*(lane>=16), N = nt*16 + (lane&15).
// Optionally sums two matrices (used to fold tj_root_w + jj_root_w).
// ---------------------------------------------------------------------------
__global__ __launch_bounds__(256) void k_pack(const float* __restrict__ W1,
                                              const float* __restrict__ W2,
                                              _Float16* __restrict__ out,
                                              int Ncols, int Ntiles, int total) {
  int idx = blockIdx.x * 256 + threadIdx.x;
  if (idx >= total) return;
  int i    = idx & 15;
  int lane = (idx >> 4) & 31;
  int t    = idx >> 9;
  int nt   = t % Ntiles;
  int kc   = t / Ntiles;
  int K = kc * 32 + i + ((lane >= 16) ? 16 : 0);
  int N = nt * 16 + (lane & 15);
  float v = W1[K * Ncols + N];
  if (W2) v += W2[K * Ncols + N];
  out[idx] = (_Float16)v;
}

__global__ __launch_bounds__(256) void k_zero(float* __restrict__ p, long long n4) {
  long long i = (long long)blockIdx.x * 256 + threadIdx.x;
  if (i < n4) {
    vf4 z = {0.f, 0.f, 0.f, 0.f};
    ((vf4*)p)[i] = z;
  }
}

// segment_sum: 16 lanes per edge, float4 gather + 4x f32 atomic scatter-add.
__global__ __launch_bounds__(256) void k_scatter(const float* __restrict__ src,
                                                 const int* __restrict__ ei, int E,
                                                 float* dst) {
  long long t = (long long)blockIdx.x * 256 + threadIdx.x;
  if (t >= (long long)E * 16) return;
  int e = (int)(t >> 4);
  int c = ((int)t & 15) * 4;
  int s = ei[e];       // row 0 of stacked [2,E]: src
  int d = ei[E + e];   // row 1: dst
  vf4 v = *(const vf4*)(src + (size_t)s * 64 + c);
  float* dp = dst + (size_t)d * 64 + c;
  atomicAdd(dp + 0, v[0]);
  atomicAdd(dp + 1, v[1]);
  atomicAdd(dp + 2, v[2]);
  atomicAdd(dp + 3, v[3]);
}

// ---------------------------------------------------------------------------
// Wave-level GEMM accumulate: acc[nt] += A[16 x 32*KCHUNKS] * Wp[.., nt*16..]
// A loaded in contiguous K order (matches packed B): lane<16 -> K {0..15}+32*kc,
// lane>=16 -> K {16..31}+32*kc, row M = lane&15.
// ---------------------------------------------------------------------------
template <int NTILES, int KCHUNKS>
__device__ __forceinline__ void gemm_acc(const float* src, int ld, int tile, int lane,
                                         const _Float16* __restrict__ wp, v8f* acc) {
  const int row = tile * 16 + (lane & 15);
  const int k0  = (lane >> 4) * 16;
  const float* rp = src + (size_t)row * ld;
  const v16h* bp = (const v16h*)wp;
#pragma unroll
  for (int kc = 0; kc < KCHUNKS; ++kc) {
    const float* p = rp + kc * 32 + k0;
    vf4 a0 = *(const vf4*)(p + 0);
    vf4 a1 = *(const vf4*)(p + 4);
    vf4 a2 = *(const vf4*)(p + 8);
    vf4 a3 = *(const vf4*)(p + 12);
    v16h A;
#pragma unroll
    for (int j = 0; j < 4; ++j) {
      A[j]      = (_Float16)a0[j];
      A[4 + j]  = (_Float16)a1[j];
      A[8 + j]  = (_Float16)a2[j];
      A[12 + j] = (_Float16)a3[j];
    }
#pragma unroll
    for (int nt = 0; nt < NTILES; ++nt) {
      v16h B = bp[(kc * NTILES + nt) * 32 + lane];
      acc[nt] = __builtin_amdgcn_wmma_f32_16x16x32_f16(
          false, A, false, B, (short)0, acc[nt], false, false);
    }
  }
}

// Input linear: out[N,64] = x[N,32] @ W + b (no activation).
__global__ __launch_bounds__(256) void k_linear(const float* __restrict__ x,
                                                const _Float16* __restrict__ wp,
                                                const float* __restrict__ b,
                                                float* __restrict__ out, int ntiles) {
  int wid  = (blockIdx.x * 256 + threadIdx.x) >> 5;
  int lane = threadIdx.x & 31;
  if (wid >= ntiles) return;  // wave-uniform: EXEC stays all-ones for WMMA
  v8f acc[4] = {{}, {}, {}, {}};
  gemm_acc<4, 1>(x, 32, wid, lane, wp, acc);
  int col16 = lane & 15, hi = lane >> 4;
#pragma unroll
  for (int nt = 0; nt < 4; ++nt) {
    float bb = b[nt * 16 + col16];
#pragma unroll
    for (int r = 0; r < 8; ++r) {
      int row = wid * 16 + r + 8 * hi;  // C/D layout: VGPR r -> M = r + 8*(lane>=16)
      out[(size_t)row * 64 + nt * 16 + col16] = acc[nt][r] + bb;
    }
  }
}

// Hetero GraphConv dense phase:
// out = tanh(msg1@Wr1 + b1 [+ msg2@Wr2 + b2] + h@Wroot).
// `out` may alias `msg1` (all loads feed WMMAs that dominate the stores).
__global__ __launch_bounds__(256) void k_dense(const float* msg1,
                                               const _Float16* __restrict__ wr1,
                                               const float* __restrict__ b1,
                                               const float* msg2,
                                               const _Float16* __restrict__ wr2,
                                               const float* __restrict__ b2,
                                               const float* hroot,
                                               const _Float16* __restrict__ wroot,
                                               float* out, int ntiles) {
  int wid  = (blockIdx.x * 256 + threadIdx.x) >> 5;
  int lane = threadIdx.x & 31;
  if (wid >= ntiles) return;
  v8f acc[4] = {{}, {}, {}, {}};
  gemm_acc<4, 2>(msg1, 64, wid, lane, wr1, acc);
  if (msg2) gemm_acc<4, 2>(msg2, 64, wid, lane, wr2, acc);
  gemm_acc<4, 2>(hroot, 64, wid, lane, wroot, acc);
  int col16 = lane & 15, hi = lane >> 4;
#pragma unroll
  for (int nt = 0; nt < 4; ++nt) {
    float bb = b1[nt * 16 + col16];
    if (msg2) bb += b2[nt * 16 + col16];
#pragma unroll
    for (int r = 0; r < 8; ++r) {
      int row = wid * 16 + r + 8 * hi;
      out[(size_t)row * 64 + nt * 16 + col16] = tanhf(acc[nt][r] + bb);
    }
  }
}

// Output head: o = tanh(h@W3 + b3); loc = o[:, :8]; scale = softplus(o[:,8:]+bias).
__global__ __launch_bounds__(256) void k_out(const float* __restrict__ h,
                                             const _Float16* __restrict__ wp,
                                             const float* __restrict__ b,
                                             float* __restrict__ out_loc,
                                             float* __restrict__ out_scale, int ntiles) {
  int wid  = (blockIdx.x * 256 + threadIdx.x) >> 5;
  int lane = threadIdx.x & 31;
  if (wid >= ntiles) return;
  v8f acc[1] = {{}};
  gemm_acc<1, 2>(h, 64, wid, lane, wp, acc);
  int col = lane & 15, hi = lane >> 4;
  float bb = b[col];
#pragma unroll
  for (int r = 0; r < 8; ++r) {
    int row = wid * 16 + r + 8 * hi;
    float v = tanhf(acc[0][r] + bb);
    if (col < 8) {
      out_loc[(size_t)row * 8 + col] = v;
    } else {
      float xin = v + kSPBias;
      float sp  = (xin > 20.f) ? xin : log1pf(expf(xin));
      out_scale[(size_t)row * 8 + (col - 8)] = sp;
    }
  }
}

// ---------------------------------------------------------------------------
extern "C" void kernel_launch(void* const* d_in, const int* in_sizes, int n_in,
                              void* d_out, int out_size, void* d_ws, size_t ws_size,
                              hipStream_t stream) {
  (void)in_sizes; (void)n_in; (void)out_size; (void)ws_size;
  const float* F = nullptr; (void)F;

  const size_t NT64 = (size_t)kNT * 64;
  const size_t NJ64 = (size_t)kNJ * 64;
  float* ws = (float*)d_ws;
  float* At = ws;                 // [NT,64]
  float* Bt = At + NT64;          // [NT,64]
  float* Aj = Bt + NT64;          // [NJ,64]
  float* Bj = Aj + NJ64;          // [NJ,64]
  float* Cj = Bj + NJ64;          // [NJ,64]
  _Float16* pk = (_Float16*)(Cj + NJ64);  // packed weights (~92 KB)
  float* dout = (float*)d_out;

  const int* ei_tj = (const int*)d_in[2];
  const int* ei_jt = (const int*)d_in[3];
  const int* ei_jj = (const int*)d_in[4];

  // --- pack weights (f32 -> f16 WMMA B layout); joint roots folded ---
  struct MatDesc { int i1, i2, K, N; };
  const MatDesc mats[14] = {
      {5, -1, 32, 64},   // 0: w0_t
      {7, -1, 32, 64},   // 1: w0_j
      {12, -1, 64, 64},  // 2: c1_jt_rel
      {14, -1, 64, 64},  // 3: c1_jt_root
      {9, -1, 64, 64},   // 4: c1_tj_rel
      {15, -1, 64, 64},  // 5: c1_jj_rel
      {11, 17, 64, 64},  // 6: c1_tj_root + c1_jj_root
      {21, -1, 64, 64},  // 7: c2_jt_rel
      {23, -1, 64, 64},  // 8: c2_jt_root
      {18, -1, 64, 64},  // 9: c2_tj_rel
      {24, -1, 64, 64},  // 10: c2_jj_rel
      {20, 26, 64, 64},  // 11: c2_tj_root + c2_jj_root
      {27, -1, 64, 16},  // 12: w3_t
      {29, -1, 64, 16},  // 13: w3_j
  };
  size_t off[14]; size_t o = 0;
  for (int m = 0; m < 14; ++m) {
    off[m] = o;
    int total = mats[m].K * mats[m].N;
    const float* w2 = (mats[m].i2 < 0) ? nullptr : (const float*)d_in[mats[m].i2];
    k_pack<<<(total + 255) / 256, 256, 0, stream>>>(
        (const float*)d_in[mats[m].i1], w2, pk + o, mats[m].N, mats[m].N / 16, total);
    o += (size_t)total;
  }

  const int TT = kNT / 16;  // 6250 tiles (exact)
  const int TJ = kNJ / 16;  // 18750 tiles (exact)
  auto gblk = [](long long threads) { return (int)((threads + 255) / 256); };

  // --- input linears ---
  k_linear<<<gblk((long long)TT * 32), 256, 0, stream>>>(
      (const float*)d_in[0], pk + off[0], (const float*)d_in[6], At, TT);
  k_linear<<<gblk((long long)TJ * 32), 256, 0, stream>>>(
      (const float*)d_in[1], pk + off[1], (const float*)d_in[8], Aj, TJ);

  // --- layer 1: h in At/Aj; messages into Bt/Bj/Cj ---
  k_zero<<<gblk(NT64 / 4), 256, 0, stream>>>(Bt, NT64 / 4);
  k_zero<<<gblk(NJ64 / 4), 256, 0, stream>>>(Bj, NJ64 / 4);
  k_zero<<<gblk(NJ64 / 4), 256, 0, stream>>>(Cj, NJ64 / 4);
  k_scatter<<<gblk(16LL * kE_JT), 256, 0, stream>>>(Aj, ei_jt, kE_JT, Bt);
  k_scatter<<<gblk(16LL * kE_TJ), 256, 0, stream>>>(At, ei_tj, kE_TJ, Bj);
  k_scatter<<<gblk(16LL * kE_JJ), 256, 0, stream>>>(Aj, ei_jj, kE_JJ, Cj);
  // torso: msg_jt@rel + b + h_t@root -> Bt (aliases msg1)
  k_dense<<<gblk((long long)TT * 32), 256, 0, stream>>>(
      Bt, pk + off[2], (const float*)d_in[13],
      nullptr, nullptr, nullptr,
      At, pk + off[3], Bt, TT);
  // joint: msg_tj@rel + msg_jj@rel + (b_tj+b_jj) + h_j@(root_tj+root_jj) -> Bj
  k_dense<<<gblk((long long)TJ * 32), 256, 0, stream>>>(
      Bj, pk + off[4], (const float*)d_in[10],
      Cj, pk + off[5], (const float*)d_in[16],
      Aj, pk + off[6], Bj, TJ);

  // --- layer 2: h in Bt/Bj; messages into At/Aj/Cj ---
  k_zero<<<gblk(NT64 / 4), 256, 0, stream>>>(At, NT64 / 4);
  k_zero<<<gblk(NJ64 / 4), 256, 0, stream>>>(Aj, NJ64 / 4);
  k_zero<<<gblk(NJ64 / 4), 256, 0, stream>>>(Cj, NJ64 / 4);
  k_scatter<<<gblk(16LL * kE_JT), 256, 0, stream>>>(Bj, ei_jt, kE_JT, At);
  k_scatter<<<gblk(16LL * kE_TJ), 256, 0, stream>>>(Bt, ei_tj, kE_TJ, Aj);
  k_scatter<<<gblk(16LL * kE_JJ), 256, 0, stream>>>(Bj, ei_jj, kE_JJ, Cj);
  k_dense<<<gblk((long long)TT * 32), 256, 0, stream>>>(
      At, pk + off[7], (const float*)d_in[22],
      nullptr, nullptr, nullptr,
      Bt, pk + off[8], At, TT);
  k_dense<<<gblk((long long)TJ * 32), 256, 0, stream>>>(
      Aj, pk + off[9], (const float*)d_in[19],
      Cj, pk + off[10], (const float*)d_in[25],
      Bj, pk + off[11], Aj, TJ);

  // --- output heads: loc_t | scale_t | loc_j | scale_j ---
  k_out<<<gblk((long long)TT * 32), 256, 0, stream>>>(
      At, pk + off[12], (const float*)d_in[28],
      dout, dout + (size_t)kNT * 8, TT);
  k_out<<<gblk((long long)TJ * 32), 256, 0, stream>>>(
      Aj, pk + off[13], (const float*)d_in[30],
      dout + (size_t)kNT * 16, dout + (size_t)kNT * 16 + (size_t)kNJ * 8, TJ);
}
